// LinearAttention_51135880626804
// MI455X (gfx1250) — matrix operand order repaired and checked by reading, not compiled, not run on hardware
//
#include <hip/hip_runtime.h>
#include <hip/hip_bf16.h>

typedef __attribute__((ext_vector_type(16))) _Float16 v16h;
typedef __attribute__((ext_vector_type(8)))  _Float16 v8h;
typedef __attribute__((ext_vector_type(8)))  float    v8f;
typedef __attribute__((ext_vector_type(4)))  unsigned int v4u;
typedef __attribute__((ext_vector_type(8)))  int          v8i;
typedef __attribute__((ext_vector_type(4)))  int          v4i;

#define SEQ  4096
#define DIMM 128
#define NB   4

// ---------------------------------------------------------------------------
// Prologue: RoPE(Q), RoPE(K) -> f16 row-major; V -> f16 transposed [B][D][S].
// ---------------------------------------------------------------------------
__global__ void rope_prep_kernel(const float* __restrict__ Q,
                                 const float* __restrict__ K,
                                 const float* __restrict__ V,
                                 _Float16* __restrict__ Qh,
                                 _Float16* __restrict__ Kh,
                                 _Float16* __restrict__ Vt) {
    int idx = blockIdx.x * blockDim.x + threadIdx.x;   // [0, B*S*D)
    int d = idx & (DIMM - 1);
    int s = (idx >> 7) & (SEQ - 1);
    int b = idx >> 19;

    float inv_freq = __powf(10000.0f, -(float)(d & 63) * (1.0f / 64.0f));
    float ang = (float)s * inv_freq;
    float c  = __cosf(ang);
    float sn = __sinf(ang);

    int   partner = (d < 64) ? idx + 64 : idx - 64;
    float sgn     = (d < 64) ? -1.0f : 1.0f;

    float q  = Q[idx];
    float k  = K[idx];
    float qr = sgn * Q[partner];
    float kr = sgn * K[partner];

    Qh[idx] = (_Float16)(q * c + qr * sn);
    Kh[idx] = (_Float16)(k * c + kr * sn);
    Vt[((size_t)b * DIMM + d) * SEQ + s] = (_Float16)V[idx];
}

// ---------------------------------------------------------------------------
// TDM: 2-D tensor_load_to_lds (D# per CDNA5 ISA).
// f16 elements (data_size code 1 = 2 bytes). Units are elements.
// ---------------------------------------------------------------------------
__device__ __forceinline__ void tdm_load_2d(unsigned int lds_off,
                                            const void* gptr,
                                            unsigned int tensor_d0,
                                            unsigned int tensor_d1,
                                            unsigned int tile_d0,
                                            unsigned int tile_d1,
                                            unsigned int stride0) {
    unsigned long long ga = (unsigned long long)gptr;
    v4u g0;
    g0.x = 1u;                                          // count=1, user mode
    g0.y = lds_off;                                     // lds_addr (bytes)
    g0.z = (unsigned int)ga;                            // global_addr[31:0]
    g0.w = (unsigned int)((ga >> 32) & 0x1FFFFFFu)      // global_addr[56:32]
         | (2u << 30);                                  // type = 2 ("image")
    v8i g1;
    g1[0] = (int)(1u << 16);                            // wg_mask=0, data_size=2B
    g1[1] = (int)((tensor_d0 & 0xFFFFu) << 16);
    g1[2] = (int)((tensor_d0 >> 16) | ((tensor_d1 & 0xFFFFu) << 16));
    g1[3] = (int)((tensor_d1 >> 16) | (tile_d0 << 16));
    g1[4] = (int)(tile_d1);                             // tile_dim2 = 0
    g1[5] = (int)stride0;
    g1[6] = 0;
    g1[7] = 0;
    v4i z4 = {0, 0, 0, 0};
    v8i z8 = {0, 0, 0, 0, 0, 0, 0, 0};
    __builtin_amdgcn_tensor_load_to_lds(g0, g1, z4, z4, z8, 0);
}

// ---------------------------------------------------------------------------
// Fused causal (Qr Kr^T) * V. First GEMM computes E^T = K_tile x Q^T so the
// masked energy repacks lane-locally into the A operand of E*V (no LDS bounce).
// ---------------------------------------------------------------------------
__launch_bounds__(256, 1)
__global__ void fused_causal_attn_kernel(const _Float16* __restrict__ Qh,
                                         const _Float16* __restrict__ Kh,
                                         const _Float16* __restrict__ Vt,
                                         float* __restrict__ O) {
    __shared__ _Float16 ktile[2][32][DIMM];   // 2 x 8KB : 32 keys x 128 d
    __shared__ _Float16 vtile[2][DIMM][32];   // 2 x 8KB : 128 d x 32 keys

    const int tid  = threadIdx.x;
    const int wave = tid >> 5;
    const int lane = tid & 31;
    const int g    = lane >> 4;
    const int r    = lane & 15;

    const int qblocks = SEQ / 128;
    const int b  = blockIdx.x / qblocks;
    const int qb = blockIdx.x % qblocks;
    const int q0 = qb * 128 + wave * 16;

    const _Float16* Qb = Qh + (size_t)b * SEQ * DIMM;
    const _Float16* Kb = Kh + (size_t)b * SEQ * DIMM;
    const _Float16* Vb = Vt + (size_t)b * DIMM * SEQ;

    // ---- Q tile as loop-invariant B-fragments of E^T = K x Q^T ----
    v16h bq[4];
    const _Float16* qrow = Qb + (size_t)(q0 + r) * DIMM;
    #pragma unroll
    for (int kc = 0; kc < 4; ++kc)
        bq[kc] = *(const v16h*)(qrow + kc * 32 + 16 * g);

    v8f o[8] = {};
    const int qidx = q0 + r;                  // this lane's query row

    // block-uniform step count (barriers must stay uniform)
    const int nsteps = (qb * 128 + 112 + 16 + 31) >> 5;

    if (wave == 0) {
        tdm_load_2d((unsigned int)(uintptr_t)&ktile[0][0][0], Kb,
                    DIMM, SEQ, DIMM, 32, DIMM);
        tdm_load_2d((unsigned int)(uintptr_t)&vtile[0][0][0], Vb,
                    SEQ, DIMM, 32, DIMM, SEQ);
    }

    for (int ks = 0; ks < nsteps; ++ks) {
        const int k0  = ks * 32;
        const int buf = ks & 1;

        if (wave == 0) __builtin_amdgcn_s_wait_tensorcnt(0);
        __syncthreads();                       // current buffers published

        if (wave == 0 && ks + 1 < nsteps) {
            const int nk0 = k0 + 32, nb_ = (ks + 1) & 1;
            tdm_load_2d((unsigned int)(uintptr_t)&ktile[nb_][0][0],
                        Kb + (size_t)nk0 * DIMM, DIMM, SEQ, DIMM, 32, DIMM);
            tdm_load_2d((unsigned int)(uintptr_t)&vtile[nb_][0][0],
                        Vb + nk0, SEQ, DIMM, 32, DIMM, SEQ);
        }

        // wave-uniform: fully-masked tile -> no work (EXEC stays all-ones)
        if (k0 <= q0 + 15) {
            // ---- batch ALL LDS fragment loads; fence so they stay batched ----
            v16h ka[8];
            #pragma unroll
            for (int kc = 0; kc < 4; ++kc) {
                const int d0 = kc * 32 + 16 * g;
                union { v16h v; v8h h[2]; } u0, u1;
                u0.h[0] = *(const v8h*)(&ktile[buf][r][d0]);
                u0.h[1] = *(const v8h*)(&ktile[buf][r][d0 + 8]);
                u1.h[0] = *(const v8h*)(&ktile[buf][16 + r][d0]);
                u1.h[1] = *(const v8h*)(&ktile[buf][16 + r][d0 + 8]);
                ka[kc]     = u0.v;
                ka[4 + kc] = u1.v;
            }
            v16h vb[8];
            #pragma unroll
            for (int dc = 0; dc < 8; ++dc) {
                union { v16h v; v8h h[2]; } uv;
                uv.h[0] = *(const v8h*)(&vtile[buf][dc * 16 + r][8 * g]);
                uv.h[1] = *(const v8h*)(&vtile[buf][dc * 16 + r][16 + 8 * g]);
                vb[dc] = uv.v;
            }

            __builtin_amdgcn_sched_barrier(0);   // loads above, compute below

            // ---- E^T tiles: two independent 4-deep wmma chains ----
            v8f e0 = {}, e1 = {};
            #pragma unroll
            for (int kc = 0; kc < 4; ++kc) {
                e0 = __builtin_amdgcn_wmma_f32_16x16x32_f16(false, ka[kc], false, bq[kc],
                                                            (short)0, e0, false, false);
                e1 = __builtin_amdgcn_wmma_f32_16x16x32_f16(false, ka[4 + kc], false, bq[kc],
                                                            (short)0, e1, false, false);
            }

            // ---- repack E^T -> A-fragment of E*V (lane-local) ----
            union { v16h v; _Float16 h[16]; } ue;
            if (k0 + 31 <= q0) {
                // interior tile: no causal masking needed
                #pragma unroll
                for (int v = 0; v < 8; ++v) {
                    ue.h[v]     = (_Float16)e0[v];
                    ue.h[v + 8] = (_Float16)e1[v];
                }
            } else {
                // diagonal tile: mask keys > q
                const int kbase = k0 + 8 * g;
                #pragma unroll
                for (int v = 0; v < 8; ++v) {
                    ue.h[v]     = (_Float16)((kbase + v)      <= qidx ? e0[v] : 0.0f);
                    ue.h[v + 8] = (_Float16)((kbase + 16 + v) <= qidx ? e1[v] : 0.0f);
                }
            }
            const v16h ea = ue.v;

            // ---- O[16x128] += E[16x32] * V[32x128]: 8 independent wmma ----
            #pragma unroll
            for (int dc = 0; dc < 8; ++dc)
                o[dc] = __builtin_amdgcn_wmma_f32_16x16x32_f16(false, ea, false, vb[dc],
                                                               (short)0, o[dc], false, false);
        }
    }

    // ---- epilogue: D tile (M=query, N=dcol) in C layout ----
    float* Ob = O + (size_t)b * SEQ * DIMM;
    #pragma unroll
    for (int dc = 0; dc < 8; ++dc) {
        #pragma unroll
        for (int v = 0; v < 8; ++v) {
            Ob[(size_t)(q0 + v + 8 * g) * DIMM + dc * 16 + r] = o[dc][v];
        }
    }
}

// ---------------------------------------------------------------------------
extern "C" void kernel_launch(void* const* d_in, const int* in_sizes, int n_in,
                              void* d_out, int out_size, void* d_ws, size_t ws_size,
                              hipStream_t stream) {
    const float* Q = (const float*)d_in[0];
    const float* K = (const float*)d_in[1];
    const float* V = (const float*)d_in[2];

    const size_t elems = (size_t)NB * SEQ * DIMM;
    _Float16* Qh = (_Float16*)d_ws;
    _Float16* Kh = Qh + elems;
    _Float16* Vt = Kh + elems;                          // 12 MB of ws

    rope_prep_kernel<<<(int)(elems / 256), 256, 0, stream>>>(Q, K, V, Qh, Kh, Vt);

    const int grid = NB * (SEQ / 128);
    fused_causal_attn_kernel<<<grid, 256, 0, stream>>>(Qh, Kh, Vt, (float*)d_out);
}
